// RPEConditionalTransformer_70987219468998
// MI455X (gfx1250) — matrix operand order repaired and checked by reading, not compiled, not run on hardware
//
#include <hip/hip_runtime.h>
#include <math.h>

// Problem constants (match reference)
constexpr int Dm = 256;   // d_model
constexpr int Hh = 4;     // heads
constexpr int DH = 64;    // per-head dim
constexpr int Np = 512;   // points

typedef __attribute__((ext_vector_type(16))) _Float16 v16h;
typedef __attribute__((ext_vector_type(8)))  _Float16 v8h;
typedef __attribute__((ext_vector_type(4)))  _Float16 v4h;
typedef __attribute__((ext_vector_type(8)))  float    v8f;
typedef __attribute__((ext_vector_type(4)))  float    f32x4;

// ---------------------------------------------------------------------------
// WMMA GEMM:  Y[M,Nc] = alpha * X[M,K] @ op(W) + bias   (optionally ReLU)
//   bt=0: op(W)=W,  W is [K,Nc] row-major (stride ldb)
//   bt=1: op(W)=W^T, W is [Nc,K] row-major (stride ldb)
// Block = 128 threads (4 waves). Block tile 64x64, wave tile 16x64.
// K chunked by 32 through DOUBLE-BUFFERED LDS: global loads for chunk k+1 are
// issued before the WMMAs of chunk k, so s_wait_loadcnt lands after the matrix
// ops. f32->f16 conversion happens at stage time (v_cvt_pk_f16_f32), fp32
// accumulation in v_wmma_f32_16x16x32_f16.
// All dims are multiples of the tiles (M%64==0, Nc%64==0, K%32==0).
// ---------------------------------------------------------------------------
__global__ __launch_bounds__(128)
void gemm_wmma(const float* __restrict__ X, const float* __restrict__ W,
               const float* __restrict__ bias, float* __restrict__ Y,
               int K, int lda, int ldb, int ldc,
               int bt, int relu, float alpha)
{
  __shared__ __align__(16) _Float16 Ash[2][64][40];   // [buf][m][k], +8 pad halfs
  __shared__ __align__(16) _Float16 Bsh[2][64][40];   // [buf][n][k] (B transposed)

  const int r0   = blockIdx.y * 64;
  const int c0   = blockIdx.x * 64;
  const int tid  = threadIdx.x;
  const int wave = tid >> 5;
  const int lane = tid & 31;
  const int l15  = lane & 15;
  const int hi   = lane >> 4;        // 0 or 1

  // staging coordinates
  const int arow  = tid >> 1;            // 0..63  (A rows / B rows when bt=1)
  const int acolh = (tid & 1) * 16;      // 0,16   (k offset)
  const int krow  = tid >> 2;            // 0..31  (k row, bt=0)
  const int nq    = (tid & 3) * 16;      // 0,16,32,48 (n offset, bt=0)

  auto loadA = [&](int k0, f32x4* r) {
    const f32x4* src = (const f32x4*)(X + (size_t)(r0 + arow) * lda + k0 + acolh);
    #pragma unroll
    for (int j = 0; j < 4; ++j) r[j] = src[j];
  };
  auto loadB = [&](int k0, f32x4* r) {
    if (bt) {
      const f32x4* src = (const f32x4*)(W + (size_t)(c0 + arow) * ldb + k0 + acolh);
      #pragma unroll
      for (int j = 0; j < 4; ++j) r[j] = src[j];
    } else {
      const f32x4* src = (const f32x4*)(W + (size_t)(k0 + krow) * ldb + c0 + nq);
      #pragma unroll
      for (int j = 0; j < 4; ++j) r[j] = src[j];
    }
  };
  auto storeA = [&](int buf, const f32x4* r) {
    #pragma unroll
    for (int j = 0; j < 4; ++j) {
      f32x4 f = r[j];
      v4h hv = { (_Float16)f.x, (_Float16)f.y, (_Float16)f.z, (_Float16)f.w };
      *(v4h*)(&Ash[buf][arow][acolh + j * 4]) = hv;
    }
  };
  auto storeB = [&](int buf, const f32x4* r) {
    if (bt) {
      #pragma unroll
      for (int j = 0; j < 4; ++j) {
        f32x4 f = r[j];
        v4h hv = { (_Float16)f.x, (_Float16)f.y, (_Float16)f.z, (_Float16)f.w };
        *(v4h*)(&Bsh[buf][arow][acolh + j * 4]) = hv;
      }
    } else {   // scatter-transpose into Bsh[n][k]
      #pragma unroll
      for (int j = 0; j < 4; ++j) {
        f32x4 f = r[j];
        Bsh[buf][nq + j * 4 + 0][krow] = (_Float16)f.x;
        Bsh[buf][nq + j * 4 + 1][krow] = (_Float16)f.y;
        Bsh[buf][nq + j * 4 + 2][krow] = (_Float16)f.z;
        Bsh[buf][nq + j * 4 + 3][krow] = (_Float16)f.w;
      }
    }
  };

  v8f acc[4] = {v8f{}, v8f{}, v8f{}, v8f{}};
  f32x4 ar[4], br[4];

  // prologue: stage chunk 0 into buffer 0
  loadA(0, ar); loadB(0, br);
  storeA(0, ar); storeB(0, br);
  __syncthreads();

  const int nchunks = K >> 5;
  for (int i = 0; i < nchunks; ++i) {
    const int  cur  = i & 1;
    const bool next = (i + 1) < nchunks;

    // issue next chunk's global loads (in flight during WMMAs below)
    if (next) { loadA((i + 1) * 32, ar); loadB((i + 1) * 32, br); }

    // ---- A fragment (ISA 16-bit A 16x32 layout) ----
    // lanes 0-15: halfs 0-7 = K 0-7,  halfs 8-15 = K 16-23
    // lanes 16-31: halfs 0-7 = K 8-15, halfs 8-15 = K 24-31
    const _Float16* ap = &Ash[cur][wave * 16 + l15][hi * 8];
    v8h alo = *(const v8h*)ap;
    v8h ahi = *(const v8h*)(ap + 16);
    v16h afrag = __builtin_shufflevector(alo, ahi,
        0,1,2,3,4,5,6,7,8,9,10,11,12,13,14,15);

    // ---- 4 B fragments + WMMA (B: N=lane&15, K same order as A fragment) ----
    #pragma unroll
    for (int c = 0; c < 4; ++c) {
      const _Float16* bp = &Bsh[cur][c * 16 + l15][hi * 8];
      v8h blo = *(const v8h*)bp;
      v8h bhi = *(const v8h*)(bp + 16);
      v16h bfrag = __builtin_shufflevector(blo, bhi,
          0,1,2,3,4,5,6,7,8,9,10,11,12,13,14,15);
      acc[c] = __builtin_amdgcn_wmma_f32_16x16x32_f16(
          false, afrag, false, bfrag, (short)0, acc[c], false, false);
    }

    // convert + store next chunk into the other buffer (waits for loads here)
    if (next) { storeA(cur ^ 1, ar); storeB(cur ^ 1, br); }
    __syncthreads();
  }

  // ---- epilogue: C/D layout — VGPR r, lane: M = r + 8*hi, N = lane&15 ----
  #pragma unroll
  for (int c = 0; c < 4; ++c) {
    int n = c0 + c * 16 + l15;
    float bv = bias ? bias[n] : 0.0f;
    int mbase = r0 + wave * 16 + hi * 8;
    #pragma unroll
    for (int r = 0; r < 8; ++r) {
      float v = acc[c][r] * alpha + bv;
      if (relu) v = fmaxf(v, 0.0f);
      Y[(size_t)(mbase + r) * ldc + n] = v;
    }
  }
}

// ---------------------------------------------------------------------------
// RPE fused score add:  S[h,n,m] += 0.125 * pos[n,m,:] . qW[h,n,:]
// (the pe-bias term q.bp is constant over m -> softmax-invariant -> dropped)
// One block per n; streams pos[n] (512x256 f32) exactly once with
// NON-TEMPORAL b128 loads: the 256 MB embedding tensors exceed the 192 MB L2,
// so caching them only evicts the reusable working set (scores/qkv/weights).
// This kernel is the HBM roofline (~1 GB total => ~45us at 23.3 TB/s).
// ---------------------------------------------------------------------------
__global__ __launch_bounds__(256)
void rpe_score_add(float* __restrict__ S, const float* __restrict__ pos,
                   const float* __restrict__ qW)
{
  __shared__ float qs[Hh][Dm];
  const int n = blockIdx.x;
  const int t = threadIdx.x;
  #pragma unroll
  for (int h = 0; h < Hh; ++h)
    qs[h][t] = qW[((size_t)h * Np + n) * Dm + t];
  __syncthreads();

  // gfx1250 prefetch of this thread's second row
  __builtin_prefetch((const void*)(pos + ((size_t)n * Np + t + 256) * Dm), 0, 0);

  #pragma unroll
  for (int mi = 0; mi < 2; ++mi) {
    int m = t + mi * 256;
    const f32x4* pp = (const f32x4*)(pos + ((size_t)n * Np + m) * Dm);
    float a0 = 0.f, a1 = 0.f, a2 = 0.f, a3 = 0.f;
    #pragma unroll 8
    for (int d4 = 0; d4 < Dm / 4; ++d4) {
      f32x4 p4 = __builtin_nontemporal_load(pp + d4);   // TH=NT stream
      int d = d4 * 4;
      a0 += p4.x*qs[0][d] + p4.y*qs[0][d+1] + p4.z*qs[0][d+2] + p4.w*qs[0][d+3];
      a1 += p4.x*qs[1][d] + p4.y*qs[1][d+1] + p4.z*qs[1][d+2] + p4.w*qs[1][d+3];
      a2 += p4.x*qs[2][d] + p4.y*qs[2][d+1] + p4.z*qs[2][d+2] + p4.w*qs[2][d+3];
      a3 += p4.x*qs[3][d] + p4.y*qs[3][d+1] + p4.z*qs[3][d+2] + p4.w*qs[3][d+3];
    }
    const float sc = 0.125f;   // 1/sqrt(DH)
    S[((size_t)0 * Np + n) * Np + m] += sc * a0;
    S[((size_t)1 * Np + n) * Np + m] += sc * a1;
    S[((size_t)2 * Np + n) * Np + m] += sc * a2;
    S[((size_t)3 * Np + n) * Np + m] += sc * a3;
  }
}

// ---------------------------------------------------------------------------
// Row softmax over 512 columns; one block per row (H*N rows).
// ---------------------------------------------------------------------------
__global__ __launch_bounds__(256)
void softmax512(float* __restrict__ S)
{
  __shared__ float red[256];
  float* row = S + (size_t)blockIdx.x * Np;
  const int t = threadIdx.x;
  float a = row[t], b = row[t + 256];

  red[t] = fmaxf(a, b);
  __syncthreads();
  for (int s = 128; s > 0; s >>= 1) { if (t < s) red[t] = fmaxf(red[t], red[t + s]); __syncthreads(); }
  float mx = red[0];
  __syncthreads();

  float ea = __expf(a - mx), eb = __expf(b - mx);
  red[t] = ea + eb;
  __syncthreads();
  for (int s = 128; s > 0; s >>= 1) { if (t < s) red[t] += red[t + s]; __syncthreads(); }
  float inv = 1.0f / red[0];

  row[t]       = ea * inv;
  row[t + 256] = eb * inv;
}

// ---------------------------------------------------------------------------
// out = LayerNorm(resid + add) * g + b ; one block per row, D=256 (1 elem/thr)
// ---------------------------------------------------------------------------
__global__ __launch_bounds__(256)
void add_layernorm(const float* __restrict__ resid, const float* __restrict__ add,
                   const float* __restrict__ g, const float* __restrict__ b,
                   float* __restrict__ out)
{
  __shared__ float red[256];
  const int row = blockIdx.x, t = threadIdx.x;
  float v = resid[(size_t)row * Dm + t] + add[(size_t)row * Dm + t];

  red[t] = v;
  __syncthreads();
  for (int s = 128; s > 0; s >>= 1) { if (t < s) red[t] += red[t + s]; __syncthreads(); }
  float mu = red[0] * (1.0f / Dm);
  __syncthreads();

  float d = v - mu;
  red[t] = d * d;
  __syncthreads();
  for (int s = 128; s > 0; s >>= 1) { if (t < s) red[t] += red[t + s]; __syncthreads(); }
  float var = red[0] * (1.0f / Dm);

  out[(size_t)row * Dm + t] = d * rsqrtf(var + 1e-5f) * g[t] + b[t];
}

// ============================ host orchestration ============================
namespace {

struct Lin { const float* w; const float* b; };
struct Blk {
  Lin q, k, v, out, expand, squeeze, p;
  const float *n1g, *n1b, *n2g, *n2b;
  bool self;
};
struct Ws { float *q, *k, *v, *hid, *tmp, *xn, *ffn, *qW, *sc; };

inline void gemm(hipStream_t s, const float* X, const float* W, const float* bias,
                 float* Y, int M, int K, int Nc, int lda, int ldb, int ldc,
                 int bt, int relu, float alpha)
{
  dim3 g(Nc / 64, M / 64);
  gemm_wmma<<<g, 128, 0, s>>>(X, W, bias, Y, K, lda, ldb, ldc, bt, relu, alpha);
}

inline void attn_tail(hipStream_t s, const Blk& B, float* x, const Ws& w)
{
  // attn out projection + add&norm + FFN + add&norm (shared by self/cross)
  gemm(s, w.hid, B.out.w, B.out.b, w.tmp, Np, Dm, Dm, Dm, Dm, Dm, 0, 0, 1.0f);
  add_layernorm<<<Np, 256, 0, s>>>(x, w.tmp, B.n1g, B.n1b, w.xn);
  gemm(s, w.xn, B.expand.w, B.expand.b, w.ffn, Np, Dm, 2 * Dm, Dm, 2 * Dm, 2 * Dm, 0, 1, 1.0f);
  gemm(s, w.ffn, B.squeeze.w, B.squeeze.b, w.tmp, Np, 2 * Dm, Dm, 2 * Dm, Dm, Dm, 0, 0, 1.0f);
  add_layernorm<<<Np, 256, 0, s>>>(w.xn, w.tmp, B.n2g, B.n2b, x);
}

inline void self_layer(hipStream_t s, const Blk& B, float* x, const float* pos, const Ws& w)
{
  gemm(s, x, B.q.w, B.q.b, w.q, Np, Dm, Dm, Dm, Dm, Dm, 0, 0, 1.0f);
  gemm(s, x, B.k.w, B.k.b, w.k, Np, Dm, Dm, Dm, Dm, Dm, 0, 0, 1.0f);
  gemm(s, x, B.v.w, B.v.b, w.v, Np, Dm, Dm, Dm, Dm, Dm, 0, 0, 1.0f);
  for (int h = 0; h < Hh; ++h) {
    // scores_h = (q_h @ k_h^T) / 8
    gemm(s, w.q + h * DH, w.k + h * DH, nullptr, w.sc + (size_t)h * Np * Np,
         Np, DH, Np, Dm, Dm, Np, 1, 0, 0.125f);
    // qW_h = q_h @ Wp_h^T   (fused RPE projection, [512,64]x[64,256])
    gemm(s, w.q + h * DH, B.p.w + h * DH, nullptr, w.qW + (size_t)h * Np * Dm,
         Np, DH, Dm, Dm, Dm, Dm, 1, 0, 1.0f);
  }
  rpe_score_add<<<Np, 256, 0, s>>>(w.sc, pos, w.qW);
  softmax512<<<Hh * Np, 256, 0, s>>>(w.sc);
  for (int h = 0; h < Hh; ++h)
    gemm(s, w.sc + (size_t)h * Np * Np, w.v + h * DH, nullptr, w.hid + h * DH,
         Np, Np, DH, Np, Dm, Dm, 0, 0, 1.0f);
  attn_tail(s, B, x, w);
}

inline void cross_layer(hipStream_t s, const Blk& B, float* x, const float* mem, const Ws& w)
{
  gemm(s, x,   B.q.w, B.q.b, w.q, Np, Dm, Dm, Dm, Dm, Dm, 0, 0, 1.0f);
  gemm(s, mem, B.k.w, B.k.b, w.k, Np, Dm, Dm, Dm, Dm, Dm, 0, 0, 1.0f);
  gemm(s, mem, B.v.w, B.v.b, w.v, Np, Dm, Dm, Dm, Dm, Dm, 0, 0, 1.0f);
  for (int h = 0; h < Hh; ++h)
    gemm(s, w.q + h * DH, w.k + h * DH, nullptr, w.sc + (size_t)h * Np * Np,
         Np, DH, Np, Dm, Dm, Np, 1, 0, 0.125f);
  softmax512<<<Hh * Np, 256, 0, s>>>(w.sc);
  for (int h = 0; h < Hh; ++h)
    gemm(s, w.sc + (size_t)h * Np * Np, w.v + h * DH, nullptr, w.hid + h * DH,
         Np, Np, DH, Np, Dm, Dm, 0, 0, 1.0f);
  attn_tail(s, B, x, w);
}

} // namespace

extern "C" void kernel_launch(void* const* d_in, const int* in_sizes, int n_in,
                              void* d_out, int out_size, void* d_ws, size_t ws_size,
                              hipStream_t stream)
{
  (void)in_sizes; (void)n_in; (void)out_size; (void)ws_size;

  const float* inF0 = (const float*)d_in[0];
  const float* inF1 = (const float*)d_in[1];
  const float* E0   = (const float*)d_in[2];
  const float* E1   = (const float*)d_in[3];

  // Params: jax pytree order — list in order, dict keys sorted, {b,w} per Lin.
  // self block keys:  expand,k,norm1_b,norm1_g,norm2_b,norm2_g,out,p,q,squeeze,v
  // cross block keys: same without p
  Blk blks[4];
  int idx = 4;
  for (int i = 0; i < 4; ++i) {
    Blk& B = blks[i];
    B.self = (i % 2 == 0);
    auto f = [&](int j) { return (const float*)d_in[idx + j]; };
    B.expand = { f(1), f(0) };
    B.k      = { f(3), f(2) };
    B.n1b = f(4); B.n1g = f(5); B.n2b = f(6); B.n2g = f(7);
    B.out    = { f(9), f(8) };
    if (B.self) {
      B.p       = { f(11), f(10) };
      B.q       = { f(13), f(12) };
      B.squeeze = { f(15), f(14) };
      B.v       = { f(17), f(16) };
      idx += 18;
    } else {
      B.p       = { nullptr, nullptr };
      B.q       = { f(11), f(10) };
      B.squeeze = { f(13), f(12) };
      B.v       = { f(15), f(14) };
      idx += 16;
    }
  }

  // Workspace carve (~11.5 MB of fp32)
  float* p = (float*)d_ws;
  float* f0 = p; p += Np * Dm;
  float* f1 = p; p += Np * Dm;
  Ws w;
  w.q   = p; p += Np * Dm;
  w.k   = p; p += Np * Dm;
  w.v   = p; p += Np * Dm;
  w.hid = p; p += Np * Dm;
  w.tmp = p; p += Np * Dm;
  w.xn  = p; p += Np * Dm;
  w.ffn = p; p += Np * 2 * Dm;
  w.qW  = p; p += (size_t)Hh * Np * Dm;
  w.sc  = p; p += (size_t)Hh * Np * Np;

  const size_t featBytes = (size_t)Np * Dm * sizeof(float);
  hipMemcpyAsync(f0, inF0, featBytes, hipMemcpyDeviceToDevice, stream);
  hipMemcpyAsync(f1, inF1, featBytes, hipMemcpyDeviceToDevice, stream);

  // BLOCKS = [self, cross, self, cross]; cross's 2nd call uses UPDATED f0.
  self_layer (stream, blks[0], f0, E0, w);
  self_layer (stream, blks[0], f1, E1, w);
  cross_layer(stream, blks[1], f0, f1, w);
  cross_layer(stream, blks[1], f1, f0, w);
  self_layer (stream, blks[2], f0, E0, w);
  self_layer (stream, blks[2], f1, E1, w);
  cross_layer(stream, blks[3], f0, f1, w);
  cross_layer(stream, blks[3], f1, f0, w);

  float* out = (float*)d_out;
  hipMemcpyAsync(out,            f0, featBytes, hipMemcpyDeviceToDevice, stream);
  hipMemcpyAsync(out + Np * Dm,  f1, featBytes, hipMemcpyDeviceToDevice, stream);
}
